// CausalSelfAttention_31525059953196
// MI455X (gfx1250) — matrix operand order repaired and checked by reading, not compiled, not run on hardware
//
#include <hip/hip_runtime.h>

// ---------------------------------------------------------------------------
// CausalSelfAttention for MI455X (gfx1250, wave32, WMMA bf16 / f32 accum)
// B=2, T=2048, D=1024, NH=16, HD=64
// ---------------------------------------------------------------------------

#define BB 2
#define TT 2048
#define DD 1024
#define NHH 16
#define HDD 64

typedef __bf16 bf16_t;
typedef bf16_t v16bf __attribute__((ext_vector_type(16)));
typedef float  v8f   __attribute__((ext_vector_type(8)));

union Frag {
    v16bf v;
    uint4 u[2];
};

#ifndef __has_builtin
#define __has_builtin(x) 0
#endif

#if __has_builtin(__builtin_amdgcn_global_load_async_to_lds_b128)
#define HAVE_ASYNC_LDS 1
#else
#define HAVE_ASYNC_LDS 0
#endif

#if HAVE_ASYNC_LDS
// Builtin signature (from clang diagnostic): first arg is
// 'int __attribute__((vector_size(16))) __device__ *'  i.e. int4 in AS1;
// LDS side is the AS3 analog.
typedef int v4i_b128 __attribute__((vector_size(16)));
typedef __attribute__((address_space(1))) v4i_b128 gv4i_t;
typedef __attribute__((address_space(3))) v4i_b128 lv4i_t;
static __device__ __forceinline__ void async_g2l_b128(const void* g, void* l) {
    __builtin_amdgcn_global_load_async_to_lds_b128((gv4i_t*)g, (lv4i_t*)l, 0, 0);
}
static __device__ __forceinline__ void wait_async0() {
#if __has_builtin(__builtin_amdgcn_s_wait_asynccnt)
    __builtin_amdgcn_s_wait_asynccnt(0);
#else
    asm volatile("s_wait_asynccnt 0" ::: "memory");
#endif
}
#endif

static __device__ __forceinline__ unsigned short f32_to_bf16(float f) {
    unsigned int u = __float_as_uint(f);
    // round-to-nearest-even
    unsigned int r = (u + 0x7FFFu + ((u >> 16) & 1u)) >> 16;
    return (unsigned short)r;
}

static __device__ __forceinline__ v8f zero8() {
    v8f z;
#pragma unroll
    for (int i = 0; i < 8; ++i) z[i] = 0.0f;
    return z;
}

static __device__ __forceinline__ v8f wmma_bf16(const Frag& a, const Frag& b, v8f c) {
    return __builtin_amdgcn_wmma_f32_16x16x32_bf16(
        /*neg_a=*/false, a.v, /*neg_b=*/false, b.v,
        /*c_mod=*/(short)0, c, /*reuse_a=*/false, /*reuse_b=*/false);
}

// ---------------------------------------------------------------------------
// elementwise f32 -> bf16
// ---------------------------------------------------------------------------
__global__ __launch_bounds__(256) void cvt_bf16_kernel(const float* __restrict__ in,
                                                       unsigned short* __restrict__ out,
                                                       int n) {
    for (int i = blockIdx.x * blockDim.x + threadIdx.x; i < n; i += gridDim.x * blockDim.x)
        out[i] = f32_to_bf16(in[i]);
}

// transpose + convert: in [R][C] f32 row-major -> out [C][R] bf16 row-major
__global__ __launch_bounds__(256) void cvt_t_bf16_kernel(const float* __restrict__ in,
                                                         unsigned short* __restrict__ out,
                                                         int R, int C) {
    int n = R * C;
    for (int i = blockIdx.x * blockDim.x + threadIdx.x; i < n; i += gridDim.x * blockDim.x) {
        int r = i / C;
        int c = i - r * C;
        out[(size_t)c * R + r] = f32_to_bf16(in[i]);
    }
}

// ---------------------------------------------------------------------------
// bf16 GEMM: C[M,N] = A[M,K] * B[K,N] + bias[N], with B supplied transposed
// as Bt[N,K].  Block tile 128x128, K-step 32, 256 threads = 8 waves,
// each wave computes 64x32 (4x2 grid of 16x16 WMMA tiles).
// Double-buffered LDS; tiles streamed with async global->LDS when available.
// ---------------------------------------------------------------------------
#define MBLK 128
#define NBLK 128
#define KSTEP 32
#define LSTR 40   // LDS row stride in bf16 elems (80 B, 16-B aligned)

template <bool OUTF32>
__global__ __launch_bounds__(256) void gemm_bf16_kernel(
    const unsigned short* __restrict__ A,   // [M][K] bf16
    const unsigned short* __restrict__ Bt,  // [N][K] bf16
    const float* __restrict__ bias,         // [N]
    void* __restrict__ Cout,                // [M][N] f32 or bf16
    int M, int N, int K) {
    __shared__ unsigned short sA[2][MBLK * LSTR];
    __shared__ unsigned short sB[2][NBLK * LSTR];

    const int tid  = threadIdx.x;
    const int lane = tid & 31;
    const int w    = tid >> 5;   // 0..7
    const int wm   = w >> 2;     // 0..1  (wave M tile)
    const int wn   = w & 3;      // 0..3  (wave N tile)
    const int r    = lane & 15;
    const int hf   = lane >> 4;
    const int m0   = blockIdx.y * MBLK;
    const int n0   = blockIdx.x * NBLK;

    v8f acc[4][2];
#pragma unroll
    for (int i = 0; i < 4; ++i)
#pragma unroll
        for (int j = 0; j < 2; ++j) acc[i][j] = zero8();

    const int nk = K / KSTEP;

    // stage one 128x32 A tile + 128x32 Bt tile into LDS buffer `buf`
    auto stage = [&](int kb, int buf) {
#pragma unroll
        for (int it = 0; it < 2; ++it) {
            int idx = tid + it * 256;           // 0..511
            int row = idx >> 2;                 // 0..127
            int ch  = (idx & 3) * 8;            // 0,8,16,24
            const unsigned short* ga = &A[(size_t)(m0 + row) * K + kb * KSTEP + ch];
            const unsigned short* gb = &Bt[(size_t)(n0 + row) * K + kb * KSTEP + ch];
#if HAVE_ASYNC_LDS
            async_g2l_b128(ga, &sA[buf][row * LSTR + ch]);
            async_g2l_b128(gb, &sB[buf][row * LSTR + ch]);
#else
            *(uint4*)&sA[buf][row * LSTR + ch] = *(const uint4*)ga;
            *(uint4*)&sB[buf][row * LSTR + ch] = *(const uint4*)gb;
#endif
        }
    };

    stage(0, 0);
    for (int kb = 0; kb < nk; ++kb) {
        const int buf = kb & 1;
#if HAVE_ASYNC_LDS
        wait_async0();          // our async writes into `buf` are done
#endif
        __syncthreads();        // everyone's writes into `buf` visible
        if (kb + 1 < nk) stage(kb + 1, buf ^ 1);   // overlap next tile with compute

        Frag fa[4], fb[2];
#pragma unroll
        for (int i = 0; i < 4; ++i) {
            const unsigned short* p = &sA[buf][(wm * 64 + i * 16 + r) * LSTR];
            // A layout: VGPR0-3 -> K = hf*8.., VGPR4-7 -> K = 16+hf*8..
            fa[i].u[0] = *(const uint4*)(p + hf * 8);
            fa[i].u[1] = *(const uint4*)(p + 16 + hf * 8);
        }
#pragma unroll
        for (int j = 0; j < 2; ++j) {
            const unsigned short* p = &sB[buf][(wn * 32 + j * 16 + r) * LSTR];
            // B layout: VGPR0-3 -> K = hf*8.., VGPR4-7 -> K = 16+hf*8..
            fb[j].u[0] = *(const uint4*)(p + hf * 8);
            fb[j].u[1] = *(const uint4*)(p + 16 + hf * 8);
        }
#pragma unroll
        for (int i = 0; i < 4; ++i)
#pragma unroll
            for (int j = 0; j < 2; ++j)
                acc[i][j] = wmma_bf16(fa[i], fb[j], acc[i][j]);
        // no trailing barrier: next iteration's top barrier orders buffer reuse
    }

    // epilogue: C layout -> lane col n = r, VGPR e -> row e + 8*hf
#pragma unroll
    for (int i = 0; i < 4; ++i)
#pragma unroll
        for (int j = 0; j < 2; ++j) {
            const int n  = n0 + wn * 32 + j * 16 + r;
            const float bv = bias[n];
#pragma unroll
            for (int e = 0; e < 8; ++e) {
                const int m = m0 + wm * 64 + i * 16 + e + 8 * hf;
                float val = acc[i][j][e] + bv;
                if (OUTF32)
                    ((float*)Cout)[(size_t)m * N + n] = val;
                else
                    ((unsigned short*)Cout)[(size_t)m * N + n] = f32_to_bf16(val);
            }
        }
}

// ---------------------------------------------------------------------------
// Flash attention: qkv bf16 [B*T][3D] (q | k | v per row), out y f32 [B*T][D]
// Workgroup = 128 thr (4 waves); each wave owns 16 query rows; WG covers 64
// queries of one (b, h).  Key loop step = 32 keys.
// ---------------------------------------------------------------------------
__global__ __launch_bounds__(128) void attn_kernel(const unsigned short* __restrict__ qkv,
                                                   float* __restrict__ y) {
    __shared__ unsigned short sV[HDD * LSTR];       // [d][key] transposed V tile
    __shared__ unsigned short sP[4][16 * LSTR];     // per-wave P tile [16][32]

    const int tid  = threadIdx.x;
    const int lane = tid & 31;
    const int w    = tid >> 5;   // 0..3
    const int r    = lane & 15;
    const int hf   = lane >> 4;
    const int b    = blockIdx.z;
    const int h    = blockIdx.y;
    const int q0   = blockIdx.x * 64;
    const int qw   = q0 + w * 16;           // this wave's first query row

    // ---- load Q fragments once (A-matrix layout, direct from global) ----
    Frag fq[2];
    {
        const size_t rowoff = ((size_t)(b * TT + qw + r)) * (3 * DD) + h * HDD;
#pragma unroll
        for (int kc = 0; kc < 2; ++kc) {
            fq[kc].u[0] = *(const uint4*)&qkv[rowoff + kc * 32 + hf * 8];
            fq[kc].u[1] = *(const uint4*)&qkv[rowoff + kc * 32 + 16 + hf * 8];
        }
    }

    float run_max[8], run_sum[8];
    v8f o[4];
#pragma unroll
    for (int j = 0; j < 8; ++j) { run_max[j] = -__builtin_inff(); run_sum[j] = 0.0f; }
#pragma unroll
    for (int t = 0; t < 4; ++t) o[t] = zero8();

    const int nkt = (q0 + 64) / 32;   // causal: keys 0 .. q0+63
    for (int kt = 0; kt < nkt; ++kt) {
        const int k0 = kt * 32;

        __syncthreads();
        // ---- stage V tile transposed: 32 keys x 64 d -> sV[d][key] ----
        for (int ii = tid; ii < 32 * 8; ii += 128) {
            int key = ii >> 3;
            int dc  = (ii & 7) * 8;
            const uint4 vv = *(const uint4*)&qkv[((size_t)(b * TT + k0 + key)) * (3 * DD) +
                                                 2 * DD + h * HDD + dc];
            const unsigned short* e = (const unsigned short*)&vv;
#pragma unroll
            for (int q = 0; q < 8; ++q) sV[(dc + q) * LSTR + key] = e[q];
        }
        __syncthreads();

        // ---- prefetch next key tile's K and V rows into cache ----
        if (kt + 1 < nkt) {
            const unsigned short* kp =
                &qkv[((size_t)(b * TT + k0 + 32 + lane)) * (3 * DD) + DD + h * HDD];
            __builtin_prefetch(kp, 0, 0);        // K row
            __builtin_prefetch(kp + DD, 0, 0);   // V row
        }

        // wave-uniform: skip tiles entirely above the causal diagonal
        if (k0 <= qw + 15) {
            const bool needmask = (k0 + 31 > qw);   // tile straddles the diagonal

            // ---- batch all K-fragment loads, then S = Q*K^T (2 tiles) ----
            Frag fk[2][2];
#pragma unroll
            for (int sk = 0; sk < 2; ++sk) {
                const size_t koff =
                    ((size_t)(b * TT + k0 + sk * 16 + r)) * (3 * DD) + DD + h * HDD;
#pragma unroll
                for (int kc = 0; kc < 2; ++kc) {
                    // B-matrix layout: VGPR0-3 K=hf*8.., VGPR4-7 K=16+hf*8..
                    fk[sk][kc].u[0] = *(const uint4*)&qkv[koff + kc * 32 + hf * 8];
                    fk[sk][kc].u[1] = *(const uint4*)&qkv[koff + kc * 32 + 16 + hf * 8];
                }
            }
            v8f st[2];
#pragma unroll
            for (int sk = 0; sk < 2; ++sk) {
                v8f c = zero8();
                c = wmma_bf16(fq[0], fk[sk][0], c);
                c = wmma_bf16(fq[1], fk[sk][1], c);
                st[sk] = c;
            }

            // ---- online softmax over these 32 keys, emit P (bf16) to LDS ----
#pragma unroll
            for (int j = 0; j < 8; ++j) {
                float s0 = st[0][j] * 0.125f;   // 1/sqrt(64)
                float s1 = st[1][j] * 0.125f;
                if (needmask) {
                    const int qi = qw + j + 8 * hf;
                    if (k0 + r > qi)      s0 = -__builtin_inff();
                    if (k0 + 16 + r > qi) s1 = -__builtin_inff();
                }
                float mx = fmaxf(s0, s1);
#pragma unroll
                for (int dm = 1; dm < 16; dm <<= 1) mx = fmaxf(mx, __shfl_xor(mx, dm, 32));
                const float mtot = fmaxf(run_max[j], mx);
                const float corr = __expf(run_max[j] - mtot);
                run_max[j] = mtot;
                const float p0 = __expf(s0 - mtot);
                const float p1 = __expf(s1 - mtot);
                float rs = p0 + p1;
#pragma unroll
                for (int dm = 1; dm < 16; dm <<= 1) rs += __shfl_xor(rs, dm, 32);
                run_sum[j] = run_sum[j] * corr + rs;
#pragma unroll
                for (int t = 0; t < 4; ++t) o[t][j] *= corr;
                const int prow = j + 8 * hf;
                sP[w][prow * LSTR + r]      = f32_to_bf16(p0);
                sP[w][prow * LSTR + 16 + r] = f32_to_bf16(p1);
            }

            // ---- reload P as A-fragment, O += P * V ----
            Frag fp;
            {
                const unsigned short* pp = &sP[w][r * LSTR];
                fp.u[0] = *(const uint4*)(pp + hf * 8);
                fp.u[1] = *(const uint4*)(pp + 16 + hf * 8);
            }
#pragma unroll
            for (int dt = 0; dt < 4; ++dt) {
                Frag fv;
                const unsigned short* pv = &sV[(dt * 16 + r) * LSTR];
                fv.u[0] = *(const uint4*)(pv + hf * 8);
                fv.u[1] = *(const uint4*)(pv + 16 + hf * 8);
                o[dt] = wmma_bf16(fp, fv, o[dt]);
            }
        }
    }

    // ---- epilogue: y[b, t, h*64 + d] = o / run_sum ----
#pragma unroll
    for (int j = 0; j < 8; ++j) {
        const int t = qw + j + 8 * hf;
        const float inv = 1.0f / run_sum[j];
#pragma unroll
        for (int dt = 0; dt < 4; ++dt)
            y[((size_t)(b * TT + t)) * DD + h * HDD + dt * 16 + r] = o[dt][j] * inv;
    }
}

// ---------------------------------------------------------------------------
// launcher
// ---------------------------------------------------------------------------
extern "C" void kernel_launch(void* const* d_in, const int* in_sizes, int n_in,
                              void* d_out, int out_size, void* d_ws, size_t ws_size,
                              hipStream_t stream) {
    (void)in_sizes; (void)n_in; (void)out_size; (void)ws_size;
    const float* x     = (const float*)d_in[0];   // [B,T,D]
    const float* Wqkv  = (const float*)d_in[1];   // [D,3D]
    const float* bqkv  = (const float*)d_in[2];   // [3D]
    const float* Wproj = (const float*)d_in[3];   // [D,D]
    const float* bproj = (const float*)d_in[4];   // [D]

    char* ws = (char*)d_ws;
    // workspace layout (bytes), total 64 MiB
    unsigned short* xb     = (unsigned short*)(ws + 0);           //  8 MiB  x bf16 [B*T][D]
    unsigned short* wqkvT  = (unsigned short*)(ws + 8388608);     //  6 MiB  Wqkv^T bf16 [3D][D]
    unsigned short* wprojT = (unsigned short*)(ws + 14680064);    //  2 MiB  Wproj^T bf16 [D][D]
    unsigned short* qkvb   = (unsigned short*)(ws + 16777216);    // 24 MiB  qkv bf16 [B*T][3D]
    float*          yf     = (float*)(ws + 41943040);             // 16 MiB  y f32 [B*T][D]
    unsigned short* yb     = (unsigned short*)(ws + 58720256);    //  8 MiB  y bf16 [B*T][D]

    const int M  = BB * TT;      // 4096
    const int N1 = 3 * DD;       // 3072
    const int K  = DD;           // 1024

    // 1) convert inputs to bf16 (weights transposed to [N][K])
    cvt_bf16_kernel<<<2048, 256, 0, stream>>>(x, xb, M * DD);
    cvt_t_bf16_kernel<<<2048, 256, 0, stream>>>(Wqkv, wqkvT, DD, N1);
    cvt_t_bf16_kernel<<<1024, 256, 0, stream>>>(Wproj, wprojT, DD, DD);

    // 2) qkv = x @ Wqkv + bqkv   (bf16 out)
    gemm_bf16_kernel<false><<<dim3(N1 / NBLK, M / MBLK), 256, 0, stream>>>(
        xb, wqkvT, bqkv, (void*)qkvb, M, N1, K);

    // 3) flash attention -> y (f32)
    attn_kernel<<<dim3(TT / 64, NHH, BB), 128, 0, stream>>>(qkvb, yf);

    // 4) y -> bf16, then out = y @ Wproj + bproj (f32 out)
    cvt_bf16_kernel<<<2048, 256, 0, stream>>>(yf, yb, M * DD);
    gemm_bf16_kernel<true><<<dim3(DD / NBLK, M / MBLK), 256, 0, stream>>>(
        yb, wprojT, bproj, d_out, M, DD, K);
}
